// VortexSPCModel_87651692577293
// MI455X (gfx1250) — compile-verified, matrix operand order checked by reading, not run on hardware
//
#include <hip/hip_runtime.h>
#include <hip/hip_bf16.h>
#include <cstdint>
#include <cstddef>

// ---------------------------------------------------------------------------
// Model constants (match reference)
// ---------------------------------------------------------------------------
#define VSZ    32000
#define DMODEL 768
#define FFN    3072
#define NH     12
#define DHEAD  64
#define NLAYER 4
#define BATCH  4
#define SEQ    1024
#define NTOK   (BATCH * SEQ)   // 4096
#define QKVDIM (3 * NH * DHEAD) // 2304

typedef __bf16 bf16;
typedef __bf16 v16bf __attribute__((ext_vector_type(16)));
typedef float  v8f   __attribute__((ext_vector_type(8)));

// ---------------------------------------------------------------------------
// WMMA helper: D = A(16x32 bf16) * B(32x16 bf16) + C(16x16 f32), wave32
// ---------------------------------------------------------------------------
__device__ __forceinline__ v8f wmma_bf16(v16bf a, v16bf b, v8f c) {
  return __builtin_amdgcn_wmma_f32_16x16x32_bf16(
      /*neg_a=*/false, a, /*neg_b=*/false, b,
      /*c_mod=*/(short)0, c, /*reuse_a=*/false, /*reuse_b=*/false);
}

// 16-bit A/B fragment from a row-major (16 x >=32) tile: element (r, k),
// r = lane&15, K octet selected by lane>>4 per the ISA A-layout:
//   halves 0..7  <- K = koff + 0..7
//   halves 8..15 <- K = koff + 16..23,  koff = (lane>>4)*8
// src must be 16B aligned per lane (ld and base multiples of 8 elements).
__device__ __forceinline__ v16bf frag_row(const bf16* __restrict__ src, int ld) {
  const int lane = threadIdx.x & 31;
  const int r    = lane & 15;
  const int koff = (lane >> 4) << 3;
  const bf16* p  = src + r * ld + koff;
  union U { float4 f4; bf16 h[8]; };
  U u0, u1;
  u0.f4 = *reinterpret_cast<const float4*>(p);
  u1.f4 = *reinterpret_cast<const float4*>(p + 16);
  v16bf out;
#pragma unroll
  for (int j = 0; j < 8; ++j) { out[j] = u0.h[j]; out[8 + j] = u1.h[j]; }
  return out;
}

// B fragment from a column-major view: element (k, r) stored at src[k*ld + r].
__device__ __forceinline__ v16bf frag_col(const bf16* __restrict__ src, int ld, int k0) {
  const int lane = threadIdx.x & 31;
  const int r    = lane & 15;
  const int koff = k0 + ((lane >> 4) << 3);
  v16bf out;
#pragma unroll
  for (int j = 0; j < 8; ++j) {
    out[j]     = src[(koff + j) * ld + r];
    out[8 + j] = src[(koff + 16 + j) * ld + r];
  }
  return out;
}

// ---------------------------------------------------------------------------
// GEMM: C[M,N] (f32) = A[M,K] (bf16, row-major) * W[N,K]^T (bf16, row-major)
// M % 128 == 0, N % 128 == 0, K % 32 == 0.
// Block: 256 threads = 8 waves; 128x128 C tile; waves arranged 4(M) x 2(N),
// each wave owns a 32x64 sub-tile (2x4 grid of 16x16 accumulators).
// Per 32-K chunk: 6 fragment loads -> 8 WMMAs per wave.
// ---------------------------------------------------------------------------
#define GLD 48  // LDS row stride (elements); multiple of 8 -> 16B aligned rows

__global__ __launch_bounds__(256) void gemm_bf16_nt(
    const bf16* __restrict__ A, const bf16* __restrict__ W,
    float* __restrict__ C, int M, int N, int K) {
  __shared__ bf16 As[128 * GLD];
  __shared__ bf16 Bs[128 * GLD];
  const int tid = threadIdx.x;
  const size_t bm = (size_t)blockIdx.y * 128;
  const size_t bn = (size_t)blockIdx.x * 128;
  const int w  = tid >> 5;
  const int wm = (w & 3) * 32;   // 0,32,64,96
  const int wn = (w >> 2) * 64;  // 0,64

  v8f acc[2][4];
  const v8f vzero = {0.f, 0.f, 0.f, 0.f, 0.f, 0.f, 0.f, 0.f};
#pragma unroll
  for (int mi = 0; mi < 2; ++mi)
#pragma unroll
    for (int j = 0; j < 4; ++j) acc[mi][j] = vzero;

  for (int k0 = 0; k0 < K; k0 += 32) {
    __syncthreads();
    // stage 128x32 A and W tiles: 512 float4 per tile, 2 per thread
#pragma unroll
    for (int i = 0; i < 2; ++i) {
      const int f = tid + 256 * i;   // 0..511
      const int r = f >> 2;          // 0..127
      const int c = (f & 3) << 3;    // 0,8,16,24
      *reinterpret_cast<float4*>(&As[r * GLD + c]) =
          *reinterpret_cast<const float4*>(&A[(bm + r) * (size_t)K + k0 + c]);
      *reinterpret_cast<float4*>(&Bs[r * GLD + c]) =
          *reinterpret_cast<const float4*>(&W[(bn + r) * (size_t)K + k0 + c]);
    }
    // prefetch next K chunk into cache (global_prefetch_b8); uniform branch
    if (k0 + 32 < K) {
      const int r = tid >> 1;
      const int c = (tid & 1) << 4;
      __builtin_prefetch(&A[(bm + r) * (size_t)K + k0 + 32 + c], 0, 1);
      __builtin_prefetch(&W[(bn + r) * (size_t)K + k0 + 32 + c], 0, 1);
    }
    __syncthreads();

    v16bf a0 = frag_row(&As[wm * GLD], GLD);
    v16bf a1 = frag_row(&As[(wm + 16) * GLD], GLD);
    v16bf b0 = frag_row(&Bs[wn * GLD], GLD);
    v16bf b1 = frag_row(&Bs[(wn + 16) * GLD], GLD);
    v16bf b2 = frag_row(&Bs[(wn + 32) * GLD], GLD);
    v16bf b3 = frag_row(&Bs[(wn + 48) * GLD], GLD);
    acc[0][0] = wmma_bf16(a0, b0, acc[0][0]);
    acc[0][1] = wmma_bf16(a0, b1, acc[0][1]);
    acc[0][2] = wmma_bf16(a0, b2, acc[0][2]);
    acc[0][3] = wmma_bf16(a0, b3, acc[0][3]);
    acc[1][0] = wmma_bf16(a1, b0, acc[1][0]);
    acc[1][1] = wmma_bf16(a1, b1, acc[1][1]);
    acc[1][2] = wmma_bf16(a1, b2, acc[1][2]);
    acc[1][3] = wmma_bf16(a1, b3, acc[1][3]);
  }

  const int lane = tid & 31;
  const int n = lane & 15, g = lane >> 4;
#pragma unroll
  for (int mi = 0; mi < 2; ++mi)
#pragma unroll
    for (int j = 0; j < 4; ++j) {
#pragma unroll
      for (int v = 0; v < 8; ++v) {
        const size_t r = bm + wm + 16 * mi + v + 8 * g;
        C[r * (size_t)N + bn + wn + 16 * j + n] = acc[mi][j][v];
      }
    }
}

// ---------------------------------------------------------------------------
// f32 -> bf16 conversion (grid-stride)
// ---------------------------------------------------------------------------
__global__ void f32_to_bf16(const float* __restrict__ in, bf16* __restrict__ out,
                            size_t n) {
  size_t i = (size_t)blockIdx.x * blockDim.x + threadIdx.x;
  const size_t stride = (size_t)gridDim.x * blockDim.x;
  for (; i < n; i += stride) out[i] = (bf16)in[i];
}

// ---------------------------------------------------------------------------
// Embedding gather + RMSNorm -> f32 h (one block per token)
// ---------------------------------------------------------------------------
__global__ __launch_bounds__(256) void embed_rmsnorm(
    const int* __restrict__ x, const float* __restrict__ E,
    const float* __restrict__ w, float* __restrict__ h) {
  __shared__ float red[256];
  const int tok = blockIdx.x;
  const size_t rowi = (size_t)x[tok] * DMODEL;
  float ss = 0.f;
  for (int i = threadIdx.x; i < DMODEL; i += 256) {
    const float v = E[rowi + i];
    ss += v * v;
  }
  red[threadIdx.x] = ss;
  __syncthreads();
  for (int s = 128; s > 0; s >>= 1) {
    if (threadIdx.x < s) red[threadIdx.x] += red[threadIdx.x + s];
    __syncthreads();
  }
  const float rs = rsqrtf(red[0] / (float)DMODEL + 1e-6f);
  for (int i = threadIdx.x; i < DMODEL; i += 256)
    h[(size_t)tok * DMODEL + i] = E[rowi + i] * rs * w[i];
}

// ---------------------------------------------------------------------------
// RMSNorm f32 row -> bf16 row (one block per token)
// ---------------------------------------------------------------------------
__global__ __launch_bounds__(256) void rmsnorm_bf16(
    const float* __restrict__ xx, const float* __restrict__ w,
    bf16* __restrict__ g) {
  __shared__ float red[256];
  const size_t base = (size_t)blockIdx.x * DMODEL;
  float ss = 0.f;
  for (int i = threadIdx.x; i < DMODEL; i += 256) {
    const float v = xx[base + i];
    ss += v * v;
  }
  red[threadIdx.x] = ss;
  __syncthreads();
  for (int s = 128; s > 0; s >>= 1) {
    if (threadIdx.x < s) red[threadIdx.x] += red[threadIdx.x + s];
    __syncthreads();
  }
  const float rs = rsqrtf(red[0] / (float)DMODEL + 1e-6f);
  for (int i = threadIdx.x; i < DMODEL; i += 256)
    g[base + i] = (bf16)(xx[base + i] * rs * w[i]);
}

// ---------------------------------------------------------------------------
// RoPE + split qkv (f32, [tok, 2304]) -> bf16 Q/K/V [B*H, T, 64]
// ---------------------------------------------------------------------------
__global__ void rope_split(const float* __restrict__ qkv, bf16* __restrict__ Q,
                           bf16* __restrict__ K, bf16* __restrict__ V) {
  const size_t gidx = (size_t)blockIdx.x * blockDim.x + threadIdx.x;
  if (gidx >= (size_t)NTOK * NH * DHEAD) return;
  const int d = (int)(gidx & 63);
  const int hh = (int)((gidx >> 6) % NH);
  const size_t ntk = gidx / (NH * DHEAD);
  const int t = (int)(ntk % SEQ);
  const int b = (int)(ntk / SEQ);
  const float* base = qkv + ntk * QKVDIM;
  const float q = base[hh * 64 + d];
  const float k = base[DMODEL + hh * 64 + d];
  const float v = base[2 * DMODEL + hh * 64 + d];
  const int j = d & 31;
  const float ang = (float)t * __powf(10000.f, -((float)(2 * j)) / 64.f);
  const float cs = __cosf(ang), sn = __sinf(ang);
  float qp, kp;
  if (d < 32) {
    qp = -base[hh * 64 + d + 32];
    kp = -base[DMODEL + hh * 64 + d + 32];
  } else {
    qp = base[hh * 64 + d - 32];
    kp = base[DMODEL + hh * 64 + d - 32];
  }
  const size_t o = ((size_t)(b * NH + hh) * SEQ + t) * DHEAD + d;
  Q[o] = (bf16)(q * cs + qp * sn);
  K[o] = (bf16)(k * cs + kp * sn);
  V[o] = (bf16)v;
}

// ---------------------------------------------------------------------------
// Flash attention: causal, SCALE = 1/8.  Block = 128 threads (4 waves).
// grid = (T/64, B*H). Wave w owns q rows [qb*64 + 16w, +16).
// Q/K/V bf16 [B*H, T, 64]; output bf16 [tok, H*64].
// ---------------------------------------------------------------------------
#define ALD 72  // LDS stride (elements), multiple of 8 -> 16B rows

__global__ __launch_bounds__(128) void attention(
    const bf16* __restrict__ Q, const bf16* __restrict__ K,
    const bf16* __restrict__ V, bf16* __restrict__ O) {
  __shared__ bf16 Ks[64 * ALD];
  __shared__ bf16 Vs[64 * ALD];
  __shared__ bf16 Ps[64 * ALD];
  const int tid = threadIdx.x;
  const int w = tid >> 5;
  const int lane = tid & 31;
  const int n = lane & 15, gh = lane >> 4;
  const int qb = blockIdx.x;
  const int bh = blockIdx.y;
  const int b = bh / NH, hh = bh % NH;

  const bf16* Qg = Q + (size_t)bh * SEQ * DHEAD;
  const bf16* Kg = K + (size_t)bh * SEQ * DHEAD;
  const bf16* Vg = V + (size_t)bh * SEQ * DHEAD;

  // Q fragments for this wave's 16 rows (held in registers for all k blocks)
  const bf16* qbase = Qg + (size_t)(qb * 64 + w * 16) * DHEAD;
  const v16bf aq0 = frag_row(qbase, DHEAD);
  const v16bf aq1 = frag_row(qbase + 32, DHEAD);

  v8f o[4];
  const v8f vzero = {0.f, 0.f, 0.f, 0.f, 0.f, 0.f, 0.f, 0.f};
#pragma unroll
  for (int t = 0; t < 4; ++t) o[t] = vzero;
  float mrow[8], lrow[8];
#pragma unroll
  for (int v = 0; v < 8; ++v) { mrow[v] = -1e30f; lrow[v] = 0.f; }

  for (int kb = 0; kb <= qb; ++kb) {
    __syncthreads();
    // stage 64x64 K and V tiles into LDS (128 threads x 4 float4 each)
    const float4* Kg4 = reinterpret_cast<const float4*>(Kg + (size_t)kb * 64 * DHEAD);
    const float4* Vg4 = reinterpret_cast<const float4*>(Vg + (size_t)kb * 64 * DHEAD);
#pragma unroll
    for (int it = 0; it < 4; ++it) {
      const int f = tid + 128 * it;     // float4 index 0..511
      const int r = f >> 3;             // row 0..63
      const int c8 = (f & 7) << 3;      // col octet
      *reinterpret_cast<float4*>(&Ks[r * ALD + c8]) = Kg4[f];
      *reinterpret_cast<float4*>(&Vs[r * ALD + c8]) = Vg4[f];
    }
    __syncthreads();

    // S = Q * K^T  (four 16x16 tiles along keys)
    v8f s[4];
#pragma unroll
    for (int t = 0; t < 4; ++t) {
      const bf16* kt = &Ks[(t * 16) * ALD];
      v8f acc = vzero;
      acc = wmma_bf16(aq0, frag_row(kt, ALD), acc);
      acc = wmma_bf16(aq1, frag_row(kt + 32, ALD), acc);
      s[t] = acc;
    }

    // scale + causal mask (mask only needed on the diagonal block)
    const float SC = 0.125f;  // 1/sqrt(64)
    if (kb == qb) {
#pragma unroll
      for (int t = 0; t < 4; ++t)
#pragma unroll
        for (int v = 0; v < 8; ++v) {
          const int qr = w * 16 + v + 8 * gh;
          const int kc = t * 16 + n;
          s[t][v] = (kc > qr) ? -1e30f : s[t][v] * SC;
        }
    } else {
#pragma unroll
      for (int t = 0; t < 4; ++t)
#pragma unroll
        for (int v = 0; v < 8; ++v) s[t][v] *= SC;
    }

    // online softmax per row (rows split: lanes 0-15 hold M=v, 16-31 M=v+8)
#pragma unroll
    for (int v = 0; v < 8; ++v) {
      float mx = fmaxf(fmaxf(s[0][v], s[1][v]), fmaxf(s[2][v], s[3][v]));
#pragma unroll
      for (int off = 8; off > 0; off >>= 1)
        mx = fmaxf(mx, __shfl_xor(mx, off, 16));
      const float mnew = fmaxf(mrow[v], mx);
      const float corr = __expf(mrow[v] - mnew);
      mrow[v] = mnew;
      float ps = 0.f;
#pragma unroll
      for (int t = 0; t < 4; ++t) {
        const float pp = __expf(s[t][v] - mnew);
        s[t][v] = pp;
        ps += pp;
      }
#pragma unroll
      for (int off = 8; off > 0; off >>= 1) ps += __shfl_xor(ps, off, 16);
      lrow[v] = lrow[v] * corr + ps;
#pragma unroll
      for (int t = 0; t < 4; ++t) o[t][v] *= corr;
    }

    // write P (bf16) to LDS so it can be re-fragmented as an A matrix
    bf16* Pw = &Ps[(w * 16) * ALD];
#pragma unroll
    for (int t = 0; t < 4; ++t)
#pragma unroll
      for (int v = 0; v < 8; ++v)
        Pw[(v + 8 * gh) * ALD + t * 16 + n] = (bf16)s[t][v];
    __syncthreads();

    // O += P * V   (V as column-major B fragments)
    const v16bf ap0 = frag_row(&Ps[(w * 16) * ALD], ALD);
    const v16bf ap1 = frag_row(&Ps[(w * 16) * ALD] + 32, ALD);
#pragma unroll
    for (int t = 0; t < 4; ++t) {
      o[t] = wmma_bf16(ap0, frag_col(&Vs[t * 16], ALD, 0), o[t]);
      o[t] = wmma_bf16(ap1, frag_col(&Vs[t * 16], ALD, 32), o[t]);
    }
  }

  // normalize and store as bf16 [tok, H*64]
#pragma unroll
  for (int v = 0; v < 8; ++v) {
    const float inv = 1.f / lrow[v];
#pragma unroll
    for (int t = 0; t < 4; ++t) o[t][v] *= inv;
  }
#pragma unroll
  for (int t = 0; t < 4; ++t)
#pragma unroll
    for (int v = 0; v < 8; ++v) {
      const size_t tok = (size_t)b * SEQ + qb * 64 + w * 16 + v + 8 * gh;
      O[tok * (NH * DHEAD) + hh * 64 + t * 16 + n] = (bf16)o[t][v];
    }
}

// ---------------------------------------------------------------------------
// Elementwise: residual add, SiLU(gate)*up -> bf16
// ---------------------------------------------------------------------------
__global__ void residual_add(float* __restrict__ h, const float* __restrict__ y,
                             size_t n) {
  const size_t i = (size_t)blockIdx.x * blockDim.x + threadIdx.x;
  if (i < n) h[i] += y[i];
}

__global__ void silu_mul(const float* __restrict__ gate,
                         const float* __restrict__ up, bf16* __restrict__ a,
                         size_t n) {
  const size_t i = (size_t)blockIdx.x * blockDim.x + threadIdx.x;
  if (i < n) {
    const float g = gate[i];
    const float s = g / (1.f + __expf(-g));
    a[i] = (bf16)(s * up[i]);
  }
}

// ---------------------------------------------------------------------------
// Host-side orchestration
// ---------------------------------------------------------------------------
static inline size_t align256(size_t b) { return (b + 255) & ~(size_t)255; }

extern "C" void kernel_launch(void* const* d_in, const int* in_sizes, int n_in,
                              void* d_out, int out_size, void* d_ws,
                              size_t ws_size, hipStream_t stream) {
  (void)in_sizes; (void)n_in; (void)out_size; (void)ws_size;
  const int*   x        = (const int*)d_in[0];
  const float* embed_w  = (const float*)d_in[1];
  const float* ln_in_w  = (const float*)d_in[2];
  const float* ln1_w    = (const float*)d_in[3];
  const float* qkv_w    = (const float*)d_in[4];
  const float* out_w    = (const float*)d_in[5];
  const float* ln2_w    = (const float*)d_in[6];
  const float* gate_w   = (const float*)d_in[7];
  const float* up_w     = (const float*)d_in[8];
  const float* down_w   = (const float*)d_in[9];
  const float* ln_out_w = (const float*)d_in[10];
  float* out = (float*)d_out;

  // ---- workspace carve ----
  char* p = (char*)d_ws;
  auto carve = [&](size_t bytes) -> char* {
    char* r = p;
    p += align256(bytes);
    return r;
  };
  const size_t nEmb  = (size_t)VSZ * DMODEL;
  const size_t nQkvW = (size_t)NLAYER * QKVDIM * DMODEL;
  const size_t nOutW = (size_t)NLAYER * DMODEL * DMODEL;
  const size_t nFfW  = (size_t)NLAYER * FFN * DMODEL;

  bf16* Ew    = (bf16*)carve(nEmb * 2);
  bf16* Wqkv  = (bf16*)carve(nQkvW * 2);
  bf16* Wout  = (bf16*)carve(nOutW * 2);
  bf16* Wg    = (bf16*)carve(nFfW * 2);
  bf16* Wu    = (bf16*)carve(nFfW * 2);
  bf16* Wd    = (bf16*)carve(nFfW * 2);
  float* h    = (float*)carve((size_t)NTOK * DMODEL * 4);
  bf16*  gbf  = (bf16*)carve((size_t)NTOK * DMODEL * 2);
  float* qkvf = (float*)carve((size_t)NTOK * QKVDIM * 4);
  bf16*  Qb   = (bf16*)carve((size_t)NTOK * DMODEL * 2);
  bf16*  Kb   = (bf16*)carve((size_t)NTOK * DMODEL * 2);
  bf16*  Vb   = (bf16*)carve((size_t)NTOK * DMODEL * 2);
  bf16*  Ob   = (bf16*)carve((size_t)NTOK * DMODEL * 2);
  float* yf   = (float*)carve((size_t)NTOK * DMODEL * 4);
  float* gatef = (float*)carve((size_t)NTOK * FFN * 4);
  float* upf   = (float*)carve((size_t)NTOK * FFN * 4);
  bf16*  abf   = (bf16*)carve((size_t)NTOK * FFN * 2);

  // ---- weight conversion to bf16 (idempotent each call) ----
  f32_to_bf16<<<2048, 256, 0, stream>>>(embed_w, Ew, nEmb);
  f32_to_bf16<<<2048, 256, 0, stream>>>(qkv_w, Wqkv, nQkvW);
  f32_to_bf16<<<2048, 256, 0, stream>>>(out_w, Wout, nOutW);
  f32_to_bf16<<<2048, 256, 0, stream>>>(gate_w, Wg, nFfW);
  f32_to_bf16<<<2048, 256, 0, stream>>>(up_w, Wu, nFfW);
  f32_to_bf16<<<2048, 256, 0, stream>>>(down_w, Wd, nFfW);

  // ---- embedding + input RMSNorm ----
  embed_rmsnorm<<<NTOK, 256, 0, stream>>>(x, embed_w, ln_in_w, h);

  const size_t nHD = (size_t)NTOK * DMODEL;     // 3,145,728
  const size_t nFF = (size_t)NTOK * FFN;        // 12,582,912

  for (int l = 0; l < NLAYER; ++l) {
    // ---- attention block ----
    rmsnorm_bf16<<<NTOK, 256, 0, stream>>>(h, ln1_w + (size_t)l * DMODEL, gbf);
    gemm_bf16_nt<<<dim3(QKVDIM / 128, NTOK / 128), 256, 0, stream>>>(
        gbf, Wqkv + (size_t)l * QKVDIM * DMODEL, qkvf, NTOK, QKVDIM, DMODEL);
    rope_split<<<(unsigned)((nHD + 255) / 256), 256, 0, stream>>>(qkvf, Qb, Kb, Vb);
    attention<<<dim3(SEQ / 64, BATCH * NH), 128, 0, stream>>>(Qb, Kb, Vb, Ob);
    gemm_bf16_nt<<<dim3(DMODEL / 128, NTOK / 128), 256, 0, stream>>>(
        Ob, Wout + (size_t)l * DMODEL * DMODEL, yf, NTOK, DMODEL, DMODEL);
    residual_add<<<(unsigned)((nHD + 255) / 256), 256, 0, stream>>>(h, yf, nHD);

    // ---- MLP block ----
    rmsnorm_bf16<<<NTOK, 256, 0, stream>>>(h, ln2_w + (size_t)l * DMODEL, gbf);
    gemm_bf16_nt<<<dim3(FFN / 128, NTOK / 128), 256, 0, stream>>>(
        gbf, Wg + (size_t)l * FFN * DMODEL, gatef, NTOK, FFN, DMODEL);
    gemm_bf16_nt<<<dim3(FFN / 128, NTOK / 128), 256, 0, stream>>>(
        gbf, Wu + (size_t)l * FFN * DMODEL, upf, NTOK, FFN, DMODEL);
    silu_mul<<<(unsigned)((nFF + 255) / 256), 256, 0, stream>>>(gatef, upf, abf, nFF);
    gemm_bf16_nt<<<dim3(DMODEL / 128, NTOK / 128), 256, 0, stream>>>(
        abf, Wd + (size_t)l * FFN * DMODEL, yf, NTOK, DMODEL, FFN);
    residual_add<<<(unsigned)((nHD + 255) / 256), 256, 0, stream>>>(h, yf, nHD);
  }

  // ---- final norm + logits ----
  rmsnorm_bf16<<<NTOK, 256, 0, stream>>>(h, ln_out_w, gbf);
  gemm_bf16_nt<<<dim3(VSZ / 128, NTOK / 128), 256, 0, stream>>>(
      gbf, Ew, out, NTOK, VSZ, DMODEL);

  // ---- second output: final hidden state h (f32) ----
  hipMemcpyAsync(out + (size_t)NTOK * VSZ, h, (size_t)NTOK * DMODEL * sizeof(float),
                 hipMemcpyDeviceToDevice, stream);
}
